// XConv_66520453481073
// MI455X (gfx1250) — compile-verified, matrix operand order checked by reading, not run on hardware
//
#include <hip/hip_runtime.h>
#include <hip/hip_bf16.h>
#include <math.h>

// ---------------- problem constants ----------------
static constexpr int N_    = 16;
static constexpr int PN_   = 2048;
static constexpr int P_    = 768;
static constexpr int KNN_  = 16;
static constexpr int DIL_  = 2;
static constexpr int CF_   = 64;
static constexpr int CPREV_= 128;
static constexpr int CIN_  = 192;   // CF + CPREV
static constexpr int DM_   = 2;
static constexpr int C_    = 256;
static constexpr int KK_   = 256;   // K*K
static constexpr int NP_   = N_ * P_;          // 12288
static constexpr int NPK_  = NP_ * KNN_;       // 196608
static constexpr float EPS_ = 1e-5f;

typedef __attribute__((ext_vector_type(16))) _Float16 v16h;
typedef __attribute__((ext_vector_type(8)))  float    v8f;

union Frag16 { v16h v; _Float16 h[16]; unsigned u[8]; };

__device__ __forceinline__ float elu_f(float x) {
    return (x > 0.0f) ? x : (__expf(x) - 1.0f);
}

// 16-byte vector load of 8 contiguous halves -> fragment quad
__device__ __forceinline__ void load8h(unsigned* dst, const _Float16* src) {
    const uint4 v = *reinterpret_cast<const uint4*>(src);
    dst[0] = v.x; dst[1] = v.y; dst[2] = v.z; dst[3] = v.w;
}
// full 16x32 fragment (two b128 loads), lane-dependent K base
__device__ __forceinline__ void load_frag(Frag16& f, const _Float16* p, int klo) {
    load8h(&f.u[0], p + klo);
    load8h(&f.u[4], p + klo + 16);
}

// ---------------- 0: zero the stats region ----------------
__global__ void zero_kernel(float* p, int n) {
    int t = blockIdx.x * blockDim.x + threadIdx.x;
    if (t < n) p[t] = 0.0f;
}

// ---------------- 1: kNN (dilated top-32, keep every 2nd) ----------------
__global__ void knn_kernel(const float* __restrict__ qrs, const float* __restrict__ pts,
                           int* __restrict__ idx) {
    int t = blockIdx.x * blockDim.x + threadIdx.x;
    if (t >= NP_) return;
    int n = t / P_, p = t % P_;
    const float* q = qrs + ((size_t)n * P_ + p) * 3;
    float qx = q[0], qy = q[1], qz = q[2];
    float bd[32]; int bi[32];
    #pragma unroll
    for (int i = 0; i < 32; ++i) { bd[i] = 3.4e38f; bi[i] = 0; }
    const float* pb = pts + (size_t)n * PN_ * 3;
    for (int j = 0; j < PN_; ++j) {
        float dx = pb[j*3+0] - qx;
        float dy = pb[j*3+1] - qy;
        float dz = pb[j*3+2] - qz;
        float d = dx*dx + dy*dy + dz*dz;
        if (d < bd[31]) {
            int pos = 31;
            while (pos > 0 && bd[pos-1] > d) { bd[pos] = bd[pos-1]; bi[pos] = bi[pos-1]; --pos; }
            bd[pos] = d; bi[pos] = j;
        }
    }
    #pragma unroll
    for (int i = 0; i < KNN_; ++i) idx[(size_t)t * KNN_ + i] = bi[i * DIL_];
}

// ---------------- 2: gather, local coords, mean-norm scaling, fts gather ----------------
// one thread per (point, k); 16-lane groups share a point -> shfl reduction.
// nn_in is stored TRANSPOSED per point: nn_inT[(pt*CIN + c)*16 + k]
__global__ void local_kernel(const float* __restrict__ pts, const float* __restrict__ qrs,
                             const float* __restrict__ fts, const int* __restrict__ idx,
                             float* __restrict__ loc, _Float16* __restrict__ loc_h,
                             _Float16* __restrict__ nn_inT) {
    int tid = blockIdx.x * blockDim.x + threadIdx.x;
    int pt = tid >> 4, k = tid & 15;
    if (pt >= NP_) return;
    int n = pt / P_, p = pt % P_;
    int j = idx[(size_t)pt * KNN_ + k];
    const float* pp = pts + ((size_t)n * PN_ + j) * 3;
    const float* qp = qrs + ((size_t)n * P_ + p) * 3;
    float lx = pp[0]-qp[0], ly = pp[1]-qp[1], lz = pp[2]-qp[2];
    float len = sqrtf(lx*lx + ly*ly + lz*lz);
    float s = len;
    s += __shfl_xor(s, 1, 16);
    s += __shfl_xor(s, 2, 16);
    s += __shfl_xor(s, 4, 16);
    s += __shfl_xor(s, 8, 16);
    float inv = 16.0f / s;                 // 1 / mean(len)
    lx *= inv; ly *= inv; lz *= inv;
    loc[(size_t)pt*48 + k*3 + 0] = lx;
    loc[(size_t)pt*48 + k*3 + 1] = ly;
    loc[(size_t)pt*48 + k*3 + 2] = lz;
    loc_h[(size_t)pt*64 + k*3 + 0] = (_Float16)lx;
    loc_h[(size_t)pt*64 + k*3 + 1] = (_Float16)ly;
    loc_h[(size_t)pt*64 + k*3 + 2] = (_Float16)lz;
    loc_h[(size_t)pt*64 + 48 + k]  = (_Float16)0.0f;   // zero K-padding 48..63
    // gather previous features into nn_inT channels 64..191 (column k)
    const float* fp = fts + ((size_t)n * PN_ + j) * CPREV_;
    _Float16* op = nn_inT + ((size_t)pt * CIN_ + CF_) * 16 + k;
    for (int c = 0; c < CPREV_; ++c) op[(size_t)c * 16] = (_Float16)fp[c];
}

// ---------------- 3: layer1  a1 = elu(loc @ w1 + b1), accumulate BN stats ----------------
__global__ void layer1_kernel(const float* __restrict__ loc, const float* __restrict__ w1,
                              const float* __restrict__ b1, float* __restrict__ a1,
                              float* __restrict__ stats) {
    __shared__ float lsum[CF_], lsq[CF_];
    for (int i = threadIdx.x; i < CF_; i += blockDim.x) { lsum[i] = 0.f; lsq[i] = 0.f; }
    __syncthreads();
    int r = blockIdx.x * blockDim.x + threadIdx.x;     // row in [0, NPK)
    if (r < NPK_) {
        int pt = r >> 4, k = r & 15;
        float l0 = loc[(size_t)pt*48 + k*3 + 0];
        float l1 = loc[(size_t)pt*48 + k*3 + 1];
        float l2 = loc[(size_t)pt*48 + k*3 + 2];
        for (int c = 0; c < CF_; ++c) {
            float v = l0*w1[c] + l1*w1[CF_ + c] + l2*w1[2*CF_ + c] + b1[c];
            v = elu_f(v);
            a1[(size_t)r*CF_ + c] = v;
            atomicAdd(&lsum[c], v);
            atomicAdd(&lsq[c], v*v);
        }
    }
    __syncthreads();
    for (int i = threadIdx.x; i < CF_; i += blockDim.x) {
        atomicAdd(&stats[i], lsum[i]);
        atomicAdd(&stats[CF_ + i], lsq[i]);
    }
}

// ---------------- 4: BN finalize -> per-channel (scale, shift) ----------------
__global__ void finalize_kernel(const float* __restrict__ stats, const float* __restrict__ gamma,
                                const float* __restrict__ beta, float* __restrict__ scsh,
                                float inv_count, int Cn) {
    int c = blockIdx.x * blockDim.x + threadIdx.x;
    if (c >= Cn) return;
    float m = stats[c] * inv_count;
    float v = stats[Cn + c] * inv_count - m*m;
    float s = gamma[c] * rsqrtf(v + EPS_);
    scsh[c] = s;
    scsh[Cn + c] = beta[c] - m * s;
}

// ---------------- 5a: normalize f32 -> f16 row-major (h1) ----------------
__global__ void norm_f16_kernel(const float* __restrict__ a, const float* __restrict__ scsh,
                                _Float16* __restrict__ out, int total, int width) {
    int t = blockIdx.x * blockDim.x + threadIdx.x;
    if (t >= total) return;
    int c = t % width;
    out[t] = (_Float16)(scsh[c] * a[t] + scsh[width + c]);
}

// ---------------- 5b: normalized h2 -> nn_inT channels 0..63 ----------------
__global__ void norm_h2_kernel(const float* __restrict__ a2, const float* __restrict__ scsh,
                               _Float16* __restrict__ nn_inT) {
    int t = blockIdx.x * blockDim.x + threadIdx.x;      // NPK * 64
    if (t >= NPK_ * CF_) return;
    int r = t >> 6, c = t & 63;
    int pt = r >> 4, k = r & 15;
    nn_inT[((size_t)pt * CIN_ + c) * 16 + k] = (_Float16)(scsh[c] * a2[t] + scsh[CF_ + c]);
}

// ---------------- 6: generic WMMA GEMM: C = elu(A@Bt^T + bias), plus BN stats ----------------
// A : (M, lda)  f16 row-major, K padded/zeroed to mult of 64
// Bt: (Nn, ldbt) f16 row-major, Bt[n][k] = B[k][n]  (N-major -> B frag == A frag loads)
// one wave computes a 16x64 output tile; blockDim = 128 (4 waves); Kk % 64 == 0.
// K-step of 64: issue all 20 b128 loads first, then 8 WMMAs -> partial loadcnt waits,
// loads of later fragments overlap earlier WMMAs.
__global__ void gemm_elu_stats_kernel(const _Float16* __restrict__ A, const _Float16* __restrict__ Bt,
                                      const float* __restrict__ bias, float* __restrict__ C,
                                      float* __restrict__ stats,
                                      int M, int Nn, int Kk, int lda, int ldbt, int ldc) {
    __shared__ float lsum[256], lsq[256];
    for (int i = threadIdx.x; i < 256; i += blockDim.x) { lsum[i] = 0.f; lsq[i] = 0.f; }
    __syncthreads();

    const int lane   = threadIdx.x & 31;
    const int wave   = threadIdx.x >> 5;
    const int nt64   = Nn >> 6;
    const int gw     = blockIdx.x * (blockDim.x >> 5) + wave;
    const int mt     = gw / nt64;
    const int nt     = gw - mt * nt64;
    const int klo    = (lane < 16) ? 0 : 8;

    if (mt * 16 < M) {
        v8f acc0 = {}, acc1 = {}, acc2 = {}, acc3 = {};
        const int row = mt * 16 + (lane & 15);
        const int cb  = nt * 64 + (lane & 15);
        const _Float16* arow  = A  + (size_t)row       * lda;
        const _Float16* brow0 = Bt + (size_t)(cb +  0) * ldbt;
        const _Float16* brow1 = Bt + (size_t)(cb + 16) * ldbt;
        const _Float16* brow2 = Bt + (size_t)(cb + 32) * ldbt;
        const _Float16* brow3 = Bt + (size_t)(cb + 48) * ldbt;
        for (int kb = 0; kb < Kk; kb += 64) {
            Frag16 a0, a1, b00, b01, b02, b03, b10, b11, b12, b13;
            load_frag(a0,  arow  + kb,      klo);
            load_frag(b00, brow0 + kb,      klo);
            load_frag(b01, brow1 + kb,      klo);
            load_frag(b02, brow2 + kb,      klo);
            load_frag(b03, brow3 + kb,      klo);
            load_frag(a1,  arow  + kb + 32, klo);
            load_frag(b10, brow0 + kb + 32, klo);
            load_frag(b11, brow1 + kb + 32, klo);
            load_frag(b12, brow2 + kb + 32, klo);
            load_frag(b13, brow3 + kb + 32, klo);
            acc0 = __builtin_amdgcn_wmma_f32_16x16x32_f16(false, a0.v, false, b00.v, (short)0, acc0, false, false);
            acc1 = __builtin_amdgcn_wmma_f32_16x16x32_f16(false, a0.v, false, b01.v, (short)0, acc1, false, false);
            acc2 = __builtin_amdgcn_wmma_f32_16x16x32_f16(false, a0.v, false, b02.v, (short)0, acc2, false, false);
            acc3 = __builtin_amdgcn_wmma_f32_16x16x32_f16(false, a0.v, false, b03.v, (short)0, acc3, false, false);
            acc0 = __builtin_amdgcn_wmma_f32_16x16x32_f16(false, a1.v, false, b10.v, (short)0, acc0, false, false);
            acc1 = __builtin_amdgcn_wmma_f32_16x16x32_f16(false, a1.v, false, b11.v, (short)0, acc1, false, false);
            acc2 = __builtin_amdgcn_wmma_f32_16x16x32_f16(false, a1.v, false, b12.v, (short)0, acc2, false, false);
            acc3 = __builtin_amdgcn_wmma_f32_16x16x32_f16(false, a1.v, false, b13.v, (short)0, acc3, false, false);
        }
        const int rbase = mt * 16 + ((lane < 16) ? 0 : 8);
        #pragma unroll
        for (int t = 0; t < 4; ++t) {
            const int col = nt * 64 + t * 16 + (lane & 15);
            const float bv = bias ? bias[col] : 0.0f;
            v8f acc = (t==0) ? acc0 : (t==1) ? acc1 : (t==2) ? acc2 : acc3;
            float cs = 0.0f, cq = 0.0f;
            #pragma unroll
            for (int r = 0; r < 8; ++r) {
                float v = elu_f(acc[r] + bv);
                C[(size_t)(rbase + r) * ldc + col] = v;
                cs += v; cq += v * v;
            }
            if (stats) {
                atomicAdd(&lsum[col & 255], cs);
                atomicAdd(&lsq[col & 255], cq);
            }
        }
    }
    __syncthreads();
    if (stats) {
        for (int i = threadIdx.x; i < Nn; i += blockDim.x) {
            atomicAdd(&stats[i],      lsum[i]);
            atomicAdd(&stats[Nn + i], lsq[i]);
        }
    }
}

// ---------------- 7: grouped x-layer: a_out[pt][g*16+m] = elu(sum_k in[k*16+g]*wx[g][m][k]) ----------------
__global__ void xlayer_kernel(const float* __restrict__ a_prev, const float* __restrict__ scsh_prev,
                              const float* __restrict__ wx, float* __restrict__ a_out,
                              float* __restrict__ stats) {
    __shared__ float lsum[KK_], lsq[KK_];
    for (int i = threadIdx.x; i < KK_; i += blockDim.x) { lsum[i] = 0.f; lsq[i] = 0.f; }
    __syncthreads();
    int tid = blockIdx.x * blockDim.x + threadIdx.x;    // NP*16 threads
    int pt = tid >> 4, g = tid & 15;
    if (pt < NP_) {
        float vin[16];
        #pragma unroll
        for (int k = 0; k < 16; ++k) {
            int q = k * 16 + g;
            vin[k] = scsh_prev[q] * a_prev[(size_t)pt * KK_ + q] + scsh_prev[KK_ + q];
        }
        const float* w = wx + g * 256;                  // wx[g][m][k]
        #pragma unroll
        for (int m = 0; m < 16; ++m) {
            float acc = 0.0f;
            #pragma unroll
            for (int k = 0; k < 16; ++k) acc += vin[k] * w[m * 16 + k];
            float v = elu_f(acc);
            int q = g * 16 + m;
            a_out[(size_t)pt * KK_ + q] = v;
            atomicAdd(&lsum[q], v);
            atomicAdd(&lsq[q], v * v);
        }
    }
    __syncthreads();
    for (int i = threadIdx.x; i < KK_; i += blockDim.x) {
        atomicAdd(&stats[i],       lsum[i]);
        atomicAdd(&stats[KK_ + i], lsq[i]);
    }
}

// ---------------- 8: per-point fts_X = X @ nn_in (WMMA) then depthwise -> d ----------------
// one wave per point; 4 points per block; fts_X staged in LDS.
// B fragments loaded 4-at-a-time into distinct registers before the WMMA group.
__global__ void fts_d_kernel(const float* __restrict__ a_x2, const float* __restrict__ scshx2,
                             const _Float16* __restrict__ nn_inT, const float* __restrict__ wdw,
                             _Float16* __restrict__ d_h) {
    __shared__ float fx[4][16 * CIN_];                  // 4 * 12 KB
    const int lane = threadIdx.x & 31;
    const int wave = threadIdx.x >> 5;
    const int pt   = blockIdx.x * 4 + wave;
    const int klo  = (lane < 16) ? 0 : 8;
    const int rb   = (lane < 16) ? 0 : 8;
    float* fxl = fx[wave];

    // A fragment: X (16x16) padded to 16x32 (K>=16 zero); vectorized f32 loads
    Frag16 a;
    const int rowa = lane & 15;
    {
        const float* av = a_x2 + (size_t)pt * KK_ + rowa * 16 + klo;
        const float* sc = scshx2 + rowa * 16 + klo;
        const float* sh = scshx2 + KK_ + rowa * 16 + klo;
        float4 f0 = *reinterpret_cast<const float4*>(av);
        float4 f1 = *reinterpret_cast<const float4*>(av + 4);
        float4 s0 = *reinterpret_cast<const float4*>(sc);
        float4 s1 = *reinterpret_cast<const float4*>(sc + 4);
        float4 h0 = *reinterpret_cast<const float4*>(sh);
        float4 h1 = *reinterpret_cast<const float4*>(sh + 4);
        a.h[0] = (_Float16)(s0.x*f0.x + h0.x);
        a.h[1] = (_Float16)(s0.y*f0.y + h0.y);
        a.h[2] = (_Float16)(s0.z*f0.z + h0.z);
        a.h[3] = (_Float16)(s0.w*f0.w + h0.w);
        a.h[4] = (_Float16)(s1.x*f1.x + h1.x);
        a.h[5] = (_Float16)(s1.y*f1.y + h1.y);
        a.h[6] = (_Float16)(s1.z*f1.z + h1.z);
        a.h[7] = (_Float16)(s1.w*f1.w + h1.w);
        #pragma unroll
        for (int j = 8; j < 16; ++j) a.h[j] = (_Float16)0.0f;
    }

    const _Float16* Bp = nn_inT + (size_t)pt * CIN_ * 16;
    auto loadB = [&](Frag16& f, int t) {
        load8h(&f.u[0], Bp + (size_t)(t * 16 + (lane & 15)) * 16 + klo);  // k = klo..klo+7
        f.u[4] = 0; f.u[5] = 0; f.u[6] = 0; f.u[7] = 0;                    // K padding 16..31
    };
    auto doTile = [&](const Frag16& f, int t) {
        v8f c = {};
        c = __builtin_amdgcn_wmma_f32_16x16x32_f16(false, a.v, false, f.v, (short)0, c, false, false);
        const int col = t * 16 + (lane & 15);
        #pragma unroll
        for (int r = 0; r < 8; ++r) fxl[(rb + r) * CIN_ + col] = c[r];
    };
    #pragma unroll
    for (int g = 0; g < 3; ++g) {
        Frag16 f0, f1, f2, f3;
        loadB(f0, 4*g + 0); loadB(f1, 4*g + 1); loadB(f2, 4*g + 2); loadB(f3, 4*g + 3);
        doTile(f0, 4*g + 0); doTile(f1, 4*g + 1); doTile(f2, 4*g + 2); doTile(f3, 4*g + 3);
    }
    __syncthreads();

    // depthwise: d[c][m] = sum_k fts_X[k][c] * wdw[c][m][k]
    #pragma unroll
    for (int i = 0; i < 6; ++i) {
        int c = lane + 32 * i;                          // 192 columns / 32 lanes
        #pragma unroll
        for (int m = 0; m < DM_; ++m) {
            const float* w = wdw + (size_t)(c * DM_ + m) * 16;
            float acc = 0.0f;
            #pragma unroll
            for (int k = 0; k < 16; ++k) acc += fxl[k * CIN_ + c] * w[k];
            d_h[(size_t)pt * (CIN_ * DM_) + c * DM_ + m] = (_Float16)acc;
        }
    }
}

// ---------------- 9: weight prep kernels (all produce N-major f16) ----------------
__global__ void f32_to_f16_kernel(const float* __restrict__ in, _Float16* __restrict__ out, int n) {
    int t = blockIdx.x * blockDim.x + threadIdx.x;
    if (t < n) out[t] = (_Float16)in[t];
}
// w2t[n][k] = w2[k][n]. w2: (64, 64)
__global__ void prep_w2t_kernel(const float* __restrict__ w2, _Float16* __restrict__ w2t) {
    int t = blockIdx.x * blockDim.x + threadIdx.x;
    if (t >= 64 * 64) return;
    int n = t >> 6, k = t & 63;
    w2t[t] = (_Float16)w2[(size_t)k * 64 + n];
}
// W0t[q][kc] = wx0[q][c][k], kc = k*3+c, cols 48..63 zero.  wx0: (256,3,16)
__global__ void prep_w0t_kernel(const float* __restrict__ wx0, _Float16* __restrict__ W0t) {
    int t = blockIdx.x * blockDim.x + threadIdx.x;      // 256*64
    if (t >= 256 * 64) return;
    int q = t >> 6, kc = t & 63;
    float v = 0.0f;
    if (kc < 48) { int k = kc / 3, c = kc - 3 * k; v = wx0[((size_t)q * 3 + c) * 16 + k]; }
    W0t[t] = (_Float16)v;
}

// ---------------- 10: final BN -> d_out ----------------
__global__ void bn_out_kernel(const float* __restrict__ a, const float* __restrict__ scsh,
                              float* __restrict__ out, int total) {
    int t = blockIdx.x * blockDim.x + threadIdx.x;
    if (t >= total) return;
    int c = t & (C_ - 1);
    out[t] = scsh[c] * a[t] + scsh[C_ + c];
}

// =====================================================================
extern "C" void kernel_launch(void* const* d_in, const int* in_sizes, int n_in,
                              void* d_out, int out_size, void* d_ws, size_t ws_size,
                              hipStream_t stream) {
    const float* pts  = (const float*)d_in[0];
    const float* fts  = (const float*)d_in[1];
    const float* qrs  = (const float*)d_in[2];
    const float* w1   = (const float*)d_in[3];
    const float* b1   = (const float*)d_in[4];
    const float* g1   = (const float*)d_in[5];
    const float* be1  = (const float*)d_in[6];
    const float* w2   = (const float*)d_in[7];
    const float* b2   = (const float*)d_in[8];
    const float* g2   = (const float*)d_in[9];
    const float* be2  = (const float*)d_in[10];
    const float* wx0  = (const float*)d_in[11];
    const float* gx0  = (const float*)d_in[12];
    const float* bx0  = (const float*)d_in[13];
    const float* wx1  = (const float*)d_in[14];
    const float* gx1  = (const float*)d_in[15];
    const float* bx1  = (const float*)d_in[16];
    const float* wx2  = (const float*)d_in[17];
    const float* gx2  = (const float*)d_in[18];
    const float* bx2  = (const float*)d_in[19];
    const float* wdw  = (const float*)d_in[20];
    const float* wpw  = (const float*)d_in[21];
    const float* gout = (const float*)d_in[22];
    const float* bout = (const float*)d_in[23];
    float* out = (float*)d_out;

    // ---- workspace layout (256B aligned), with reuse ----
    char* base = (char*)d_ws;
    size_t off = 0;
    auto alloc = [&](size_t bytes) -> char* {
        off = (off + 255) & ~(size_t)255;
        char* p = base + off;
        off += bytes;
        return p;
    };
    int*       ws_idx   = (int*)      alloc((size_t)NP_ * KNN_ * 4);
    float*     ws_loc   = (float*)    alloc((size_t)NP_ * 48 * 4);
    _Float16*  ws_loch  = (_Float16*) alloc((size_t)NP_ * 64 * 2);
    float*     ws_a1    = (float*)    alloc((size_t)NPK_ * CF_ * 4);   // reused as a2
    _Float16*  ws_h1h   = (_Float16*) alloc((size_t)NPK_ * CF_ * 2);   // reused as d_h
    _Float16*  ws_nnin  = (_Float16*) alloc((size_t)NP_ * CIN_ * KNN_ * 2);  // transposed (pt,c,k)
    _Float16*  ws_w2t   = (_Float16*) alloc((size_t)CF_ * CF_ * 2);
    _Float16*  ws_W0t   = (_Float16*) alloc((size_t)256 * 64 * 2);
    _Float16*  ws_wpwh  = (_Float16*) alloc((size_t)256 * 384 * 2);
    float*     ws_ax0   = (float*)    alloc((size_t)NP_ * KK_ * 4);    // reused as a_x2
    float*     ws_ax1   = (float*)    alloc((size_t)NP_ * KK_ * 4);    // reused as a_out
    float*     ws_stats = (float*)    alloc(2304 * 4);
    float*     ws_scsh  = (float*)    alloc(2304 * 4);
    float*     ws_a2    = ws_a1;
    _Float16*  ws_dh    = ws_h1h;
    float*     ws_ax2   = ws_ax0;
    float*     ws_aout  = ws_ax1;
    // stats / scsh sub-layout (float offsets)
    float *s1 = ws_stats,        *s2 = ws_stats + 128,  *sx0 = ws_stats + 256;
    float *sx1 = ws_stats + 768, *sx2 = ws_stats + 1280, *sout = ws_stats + 1792;
    float *c1 = ws_scsh,         *c2 = ws_scsh + 128,   *cx0 = ws_scsh + 256;
    float *cx1 = ws_scsh + 768,  *cx2 = ws_scsh + 1280,  *cout = ws_scsh + 1792;

    // ---- 0: zero stats ----
    zero_kernel<<<(2304 + 255) / 256, 256, 0, stream>>>(ws_stats, 2304);

    // ---- weight preps (independent) ----
    prep_w2t_kernel<<<(CF_*CF_ + 255) / 256, 256, 0, stream>>>(w2, ws_w2t);
    prep_w0t_kernel<<<(256*64 + 255) / 256, 256, 0, stream>>>(wx0, ws_W0t);
    f32_to_f16_kernel<<<(256*384 + 255) / 256, 256, 0, stream>>>(wpw, ws_wpwh, 256*384);

    // ---- 1: kNN ----
    knn_kernel<<<(NP_ + 255) / 256, 256, 0, stream>>>(qrs, pts, ws_idx);

    // ---- 2: local coords + fts gather (transposed nn_in) ----
    local_kernel<<<(NPK_ + 255) / 256, 256, 0, stream>>>(pts, qrs, fts, ws_idx,
                                                         ws_loc, ws_loch, ws_nnin);

    // ---- 3: layer1 + stats ----
    layer1_kernel<<<(NPK_ + 255) / 256, 256, 0, stream>>>(ws_loc, w1, b1, ws_a1, s1);
    finalize_kernel<<<1, 64, 0, stream>>>(s1, g1, be1, c1, 1.0f / (float)NPK_, CF_);
    norm_f16_kernel<<<((NPK_*CF_) + 255) / 256, 256, 0, stream>>>(ws_a1, c1, ws_h1h,
                                                                  NPK_*CF_, CF_);

    // ---- 4: layer2 (WMMA): M=196608, N=64, K=64 ----
    gemm_elu_stats_kernel<<<NPK_/16/4, 128, 0, stream>>>(ws_h1h, ws_w2t, b2, ws_a2, s2,
                                                         NPK_, CF_, 64, CF_, 64, CF_);
    finalize_kernel<<<1, 64, 0, stream>>>(s2, g2, be2, c2, 1.0f / (float)NPK_, CF_);
    // normalized h2 -> nn_inT channels 0..63
    norm_h2_kernel<<<((NPK_*CF_) + 255) / 256, 256, 0, stream>>>(ws_a2, c2, ws_nnin);

    // ---- 5: x0 (WMMA): M=12288, N=256, K=48->64 ----
    gemm_elu_stats_kernel<<<(NP_/16)*4/4, 128, 0, stream>>>(ws_loch, ws_W0t, nullptr, ws_ax0, sx0,
                                                            NP_, 256, 64, 64, 64, 256);
    finalize_kernel<<<1, 256, 0, stream>>>(sx0, gx0, bx0, cx0, 1.0f / (float)NP_, KK_);

    // ---- 6: x1, x2 grouped layers ----
    xlayer_kernel<<<(NP_*16) / 256, 256, 0, stream>>>(ws_ax0, cx0, wx1, ws_ax1, sx1);
    finalize_kernel<<<1, 256, 0, stream>>>(sx1, gx1, bx1, cx1, 1.0f / (float)NP_, KK_);
    xlayer_kernel<<<(NP_*16) / 256, 256, 0, stream>>>(ws_ax1, cx1, wx2, ws_ax2, sx2);
    finalize_kernel<<<1, 256, 0, stream>>>(sx2, gx2, bx2, cx2, 1.0f / (float)NP_, KK_);

    // ---- 7: per-point fts_X (WMMA) + depthwise -> d ----
    fts_d_kernel<<<NP_ / 4, 128, 0, stream>>>(ws_ax2, cx2, ws_nnin, wdw, ws_dh);

    // ---- 8: final pointwise (WMMA): M=12288, N=256, K=384 ----
    gemm_elu_stats_kernel<<<(NP_/16)*4/4, 128, 0, stream>>>(ws_dh, ws_wpwh, nullptr, ws_aout, sout,
                                                            NP_, 256, 384, 384, 384, 256);
    finalize_kernel<<<1, 256, 0, stream>>>(sout, gout, bout, cout, 1.0f / (float)NP_, C_);

    // ---- 9: final BN -> output ----
    bn_out_kernel<<<(NP_*C_) / 256, 256, 0, stream>>>(ws_aout, cout, out, NP_*C_);
}
